// Model_14001593385050
// MI455X (gfx1250) — matrix operand order repaired and checked by reading, not compiled, not run on hardware
//
#include <hip/hip_runtime.h>
#include <math.h>

typedef __attribute__((ext_vector_type(16))) _Float16 v16h;
typedef __attribute__((ext_vector_type(8)))  float    v8f;
typedef int v4i_vs __attribute__((vector_size(4 * sizeof(int))));

#define NNODE 10000
#define NEDGE 256000
#define FDIM  512                      // MUL*16
#define RMAXF 3.5f
#define INV_SQRT_NAVG 0.19764235376052367f   // 1/sqrt(25.6)
#define INV_SQRT_MUL  0.17677669529663687f   // 1/sqrt(32)
#define INV_SQRT_NB   0.31622776601683794f   // 1/sqrt(10)

#if __has_builtin(__builtin_amdgcn_global_load_async_to_lds_b128) && \
    __has_builtin(__builtin_amdgcn_s_wait_asynccnt)
#define HAVE_ASYNC 1
#else
#define HAVE_ASYNC 0
#endif

union U16h { v16h v; _Float16 h[16]; };
union U8f  { v8f  v; float    f[8];  };
union Hq   { float4 q; _Float16 h[8]; };

__device__ __forceinline__ float fast_rcp(float x) { return __builtin_amdgcn_rcpf(x); }
__device__ __forceinline__ float silu_f(float v)   { return v * fast_rcp(1.f + __expf(-v)); }
__device__ __forceinline__ float sigm_f(float v)   { return fast_rcp(1.f + __expf(-v)); }

// 16-byte global -> LDS copy. Async (ASYNCcnt) when the CDNA5 builtin exists;
// plain register copy otherwise (also what the host pass sees).
__device__ __forceinline__ void async_ld16(const void* g, void* l) {
#if HAVE_ASYNC
  __builtin_amdgcn_global_load_async_to_lds_b128(
      (v4i_vs*)(uintptr_t)g, (v4i_vs*)l, 0, 0);
#else
  *(float4*)l = *(const float4*)g;
#endif
}
__device__ __forceinline__ void async_wait_all() {
#if HAVE_ASYNC
  __builtin_amdgcn_s_wait_asynccnt(0);
#endif
}

// ---------------------------------------------------------------------------
// Edge geometry: spherical harmonics * cutoff -> eattr[E][16] (f32),
// Gaussian radial embedding -> embh[E][16] (f16, zero-padded cols 10..15).
// ---------------------------------------------------------------------------
__global__ __launch_bounds__(256) void geom_kernel(
    const float* __restrict__ pos, const int* __restrict__ esrc,
    const int* __restrict__ edst, float* __restrict__ eattr,
    _Float16* __restrict__ embh)
{
  int e = blockIdx.x * 256 + threadIdx.x;
  if (e >= NEDGE) return;
  int s = esrc[e], d = edst[e];
  float vx = pos[s*3+0] - pos[d*3+0];
  float vy = pos[s*3+1] - pos[d*3+1];
  float vz = pos[s*3+2] - pos[d*3+2];
  float r = sqrtf(vx*vx + vy*vy + vz*vz);
  float inv = 1.f / (r + 1e-9f);
  float x = vx*inv, y = vy*inv, z = vz*inv;
  const float s3  = 1.7320508075688772f;   // sqrt(3)
  const float s15 = 3.8729833462074170f;   // sqrt(15)
  const float s5  = 2.2360679774997900f;   // sqrt(5)
  const float ca  = 2.0916500663351889f;   // sqrt(35/8)
  const float cb  = 10.246950765959598f;   // sqrt(105)
  const float cc  = 1.6201851746019651f;   // sqrt(21/8)
  const float cd  = 1.3228756555322954f;   // sqrt(7)/2
  float x2 = x*x, y2 = y*y, z2 = z*z;
  float sh[16];
  sh[0]  = 1.f;
  sh[1]  = s3*x;  sh[2] = s3*y;  sh[3] = s3*z;
  sh[4]  = s15*x*y;
  sh[5]  = s15*y*z;
  sh[6]  = 0.5f*s5*(3.f*z2 - 1.f);
  sh[7]  = s15*x*z;
  sh[8]  = 0.5f*s15*(x2 - y2);
  sh[9]  = ca*y*(3.f*x2 - y2);
  sh[10] = cb*x*y*z;
  sh[11] = cc*y*(5.f*z2 - 1.f);
  sh[12] = cd*z*(5.f*z2 - 3.f);
  sh[13] = cc*x*(5.f*z2 - 1.f);
  sh[14] = 0.5f*cb*z*(x2 - y2);
  sh[15] = ca*x*(x2 - 3.f*y2);
  float t   = fminf(fmaxf(r * (1.f/RMAXF), 0.f), 1.f);
  float cut = (r < RMAXF) ? 0.5f*(__cosf(3.14159265358979f*t) + 1.f) : 0.f;
  float* ep = eattr + (size_t)e*16;
  #pragma unroll
  for (int i = 0; i < 16; ++i) ep[i] = cut * sh[i];
  const float step  = RMAXF / 9.f;
  const float istep = 9.f / RMAXF;
  #pragma unroll
  for (int j = 0; j < 16; ++j) {
    float v = 0.f;
    if (j < 10) { float q = (r - step*(float)j) * istep; v = __expf(-q*q) * 2.8234621965697562f; }
    embh[(size_t)e*16 + j] = (_Float16)v;
  }
}

// ---------------------------------------------------------------------------
// Weight pre-packing into WMMA B-fragment lane order (f16, zero-padded).
// ---------------------------------------------------------------------------
__global__ __launch_bounds__(256) void pack_w1_kernel(
    const float* __restrict__ W1, _Float16* __restrict__ frag)
{
  int idx = blockIdx.x * 256 + threadIdx.x;
  if (idx >= 2048) return;
  int g = idx >> 9, r = idx & 511, lane = r >> 4, i = r & 15;
  int hi = lane >> 4, col = lane & 15;
  int k = (i < 8) ? (8*hi + i) : (16 + 8*hi + (i - 8)), n = g*16 + col;
  frag[idx] = (k < 10) ? (_Float16)W1[k*64 + n] : (_Float16)0.f;
}

__global__ __launch_bounds__(256) void pack_w2_kernel(
    const float* __restrict__ W2, _Float16* __restrict__ frag, int OUTC)
{
  int total = (OUTC/16) * 1024;
  int idx = blockIdx.x * 256 + threadIdx.x;
  if (idx >= total) return;
  int ct = idx >> 10, rem = idx & 1023, kk = rem >> 9, r = rem & 511;
  int lane = r >> 4, i = r & 15;
  int hi = lane >> 4, col = lane & 15;
  int k = kk*32 + ((i < 8) ? (8*hi + i) : (16 + 8*hi + (i - 8))), n = ct*16 + col;
  frag[idx] = (_Float16)W2[k*OUTC + n];
}

// ---------------------------------------------------------------------------
// Fused radial MLP for a 16-edge tile via WMMA (128 threads = 4 waves).
// A/B/C VGPR layouts per CDNA5 ISA 7.12.2 (wave32).
// ---------------------------------------------------------------------------
template <int OUTC>
__device__ __forceinline__ void radial_wmma(
    const _Float16* __restrict__ embh, int e0,
    const _Float16* __restrict__ W1f, const _Float16* __restrict__ W2f,
    _Float16 (*h_lds)[64], float (*w_lds)[OUTC])
{
  int tid = threadIdx.x, lane = tid & 31, wv = tid >> 5;
  int row = lane & 15, hi = lane >> 4, col = lane & 15;

  // --- GEMM1: one 16x16 tile of h per wave (cols wv*16..wv*16+15) ---
  U16h a;
  {
    Hq t; t.q = *(const float4*)(embh + (size_t)(e0 + row)*16 + 8*hi);
    #pragma unroll
    for (int i = 0; i < 8; ++i) { a.h[i] = t.h[i]; a.h[8 + i] = (_Float16)0.f; }
  }
  U16h b;
  b.v = *(const v16h*)(W1f + wv*512 + lane*16);
  v8f c = {};
  c = __builtin_amdgcn_wmma_f32_16x16x32_f16(false, a.v, false, b.v, (short)0, c, false, false);
  U8f cu; cu.v = c;
  #pragma unroll
  for (int j = 0; j < 8; ++j) {                    // C/D: row j+8*hi, col lane&15
    float v = cu.f[j] * INV_SQRT_NB;
    h_lds[j + 8*hi][wv*16 + col] = (_Float16)silu_f(v);
  }
  __syncthreads();

  // --- GEMM2: K=64 as two K=32 WMMAs; OUTC/64 col-tiles per wave ---
  U16h a0, a1;
  {
    Hq t;
    t.q = *(const float4*)(&h_lds[row][8*hi]);
    #pragma unroll
    for (int i = 0; i < 8; ++i) a0.h[i] = t.h[i];
    t.q = *(const float4*)(&h_lds[row][16 + 8*hi]);
    #pragma unroll
    for (int i = 0; i < 8; ++i) a0.h[8 + i] = t.h[i];
    t.q = *(const float4*)(&h_lds[row][32 + 8*hi]);
    #pragma unroll
    for (int i = 0; i < 8; ++i) a1.h[i] = t.h[i];
    t.q = *(const float4*)(&h_lds[row][48 + 8*hi]);
    #pragma unroll
    for (int i = 0; i < 8; ++i) a1.h[8 + i] = t.h[i];
  }
  const int PW = (OUTC / 16) / 4;                  // col tiles per wave
  #pragma unroll
  for (int t = 0; t < PW; ++t) {
    int ct = wv*PW + t;
    U16h b0, b1;
    b0.v = *(const v16h*)(W2f + ct*1024 + lane*16);
    b1.v = *(const v16h*)(W2f + ct*1024 + 512 + lane*16);
    v8f acc = {};
    acc = __builtin_amdgcn_wmma_f32_16x16x32_f16(false, a0.v, false, b0.v, (short)0, acc, false, false);
    acc = __builtin_amdgcn_wmma_f32_16x16x32_f16(false, a1.v, false, b1.v, (short)0, acc, false, false);
    U8f au; au.v = acc;
    #pragma unroll
    for (int j = 0; j < 8; ++j) w_lds[j + 8*hi][ct*16 + col] = au.f[j] * 0.125f;
  }
}

// ---------------------------------------------------------------------------
// Layer message kernel: 16 edges/block. eattr tile + edge indices staged in
// LDS (async copy overlapped with the radial WMMA), then (edge,channel) lanes
// compute TP messages and atomically scatter into agg.
// ---------------------------------------------------------------------------
__global__ __launch_bounds__(128) void layer_msg_kernel(
    const float* __restrict__ feat, const int* __restrict__ esrc,
    const int* __restrict__ edst, const float* __restrict__ eattr,
    const _Float16* __restrict__ embh, const _Float16* __restrict__ W1f,
    const _Float16* __restrict__ W2f, float* __restrict__ agg)
{
  __shared__ _Float16 h_lds[16][64];
  __shared__ float    w_lds[16][256];
  __shared__ float    ea_lds[16][16];
  __shared__ int      s_lds[16], d_lds[16];
  int tid = threadIdx.x;
  int e0 = blockIdx.x * 16;

  if (tid < 64) async_ld16((const char*)(eattr + (size_t)e0*16) + tid*16,
                           (char*)&ea_lds[0][0] + tid*16);
  if (tid < 16) { s_lds[tid] = esrc[e0 + tid]; d_lds[tid] = edst[e0 + tid]; }

  radial_wmma<256>(embh, e0, W1f, W2f, h_lds, w_lds);
  async_wait_all();
  __syncthreads();

  #pragma unroll
  for (int it = 0; it < 4; ++it) {
    int idx = it*128 + tid;
    int el = idx >> 5, cch = idx & 31;
    int s = s_lds[el], d = d_lds[el];
    const float4* xs4 = (const float4*)(feat + (size_t)s*FDIM + cch*16);
    float4 X0 = xs4[0], X1 = xs4[1], X2 = xs4[2], X3 = xs4[3];
    float xs[16] = {X0.x,X0.y,X0.z,X0.w, X1.x,X1.y,X1.z,X1.w,
                    X2.x,X2.y,X2.z,X2.w, X3.x,X3.y,X3.z,X3.w};
    const float4* ea4 = (const float4*)&ea_lds[el][0];
    float4 E0 = ea4[0], E1 = ea4[1], E2 = ea4[2], E3 = ea4[3];
    float ea[16] = {E0.x,E0.y,E0.z,E0.w, E1.x,E1.y,E1.z,E1.w,
                    E2.x,E2.y,E2.z,E2.w, E3.x,E3.y,E3.z,E3.w};
    const float* wl = &w_lds[el][0];       // [2][4][32] flattened
    float msg[16];
    float scal = wl[cch] * xs[0] * ea[0];  // w[0,0,c]
    { // l=1, slice [1,4)
      float dt = xs[1]*ea[1] + xs[2]*ea[2] + xs[3]*ea[3];
      scal += wl[128 + 32 + cch] * dt;
      float g = wl[32 + cch] * xs[0];
      msg[1] = g*ea[1]; msg[2] = g*ea[2]; msg[3] = g*ea[3];
    }
    { // l=2, slice [4,9)
      float dt = 0.f;
      #pragma unroll
      for (int i = 4; i < 9; ++i) dt += xs[i]*ea[i];
      scal += wl[128 + 64 + cch] * dt;
      float g = wl[64 + cch] * xs[0];
      #pragma unroll
      for (int i = 4; i < 9; ++i) msg[i] = g*ea[i];
    }
    { // l=3, slice [9,16)
      float dt = 0.f;
      #pragma unroll
      for (int i = 9; i < 16; ++i) dt += xs[i]*ea[i];
      scal += wl[128 + 96 + cch] * dt;
      float g = wl[96 + cch] * xs[0];
      #pragma unroll
      for (int i = 9; i < 16; ++i) msg[i] = g*ea[i];
    }
    msg[0] = scal;
    float* ag = agg + (size_t)d*FDIM + cch*16;
    #pragma unroll
    for (int i = 0; i < 16; ++i) atomicAdd(&ag[i], msg[i] * INV_SQRT_NAVG);
  }
}

// ---------------------------------------------------------------------------
// Self-connection + gate: 8 nodes/block, feat & Wsc staged in LDS.
// ---------------------------------------------------------------------------
__global__ __launch_bounds__(256) void sc_gate_kernel(
    const float* __restrict__ feat, const float* __restrict__ agg,
    const float* __restrict__ Wsc, float* __restrict__ out)
{
  __shared__ float wsc[4096];            // [4][32][32]
  __shared__ float fl[8][32][16];
  int tid = threadIdx.x;
  int n0 = blockIdx.x * 8;
  for (int i = tid; i < 4096; i += 256) wsc[i] = Wsc[i];
  for (int i = tid; i < 4096; i += 256) ((float*)fl)[i] = feat[(size_t)n0*FDIM + i];
  __syncthreads();
  int nl = tid >> 5, o = tid & 31;
  int n = n0 + nl;
  const float* ag = agg + (size_t)n*FDIM + o*16;
  float y[16];
  #pragma unroll
  for (int i = 0; i < 16; ++i) {
    int l = (i == 0) ? 0 : (i < 4) ? 1 : (i < 9) ? 2 : 3;
    float acc = 0.f;
    #pragma unroll
    for (int c2 = 0; c2 < 32; ++c2) acc += fl[nl][c2][i] * wsc[l*1024 + c2*32 + o];
    y[i] = acc * INV_SQRT_MUL + ag[i];
  }
  float sg = sigm_f(y[0]);
  float* op = out + (size_t)n*FDIM + o*16;
  op[0] = y[0] * sg;
  #pragma unroll
  for (int i = 1; i < 16; ++i) op[i] = y[i] * sg;
}

// ---------------------------------------------------------------------------
// Final edge readout: radial WMMA (OUTC=128), per-edge wave reduction over
// channels, one atomic per edge into node[].
// ---------------------------------------------------------------------------
__global__ __launch_bounds__(128) void final_msg_kernel(
    const float* __restrict__ feat, const int* __restrict__ esrc,
    const int* __restrict__ edst, const float* __restrict__ eattr,
    const _Float16* __restrict__ embh, const _Float16* __restrict__ W1f,
    const _Float16* __restrict__ W2f, float* __restrict__ node)
{
  __shared__ _Float16 h_lds[16][64];
  __shared__ float    w_lds[16][128];
  __shared__ float    ea_lds[16][16];
  __shared__ int      s_lds[16], d_lds[16];
  int tid = threadIdx.x, lane = tid & 31;
  int e0 = blockIdx.x * 16;

  if (tid < 64) async_ld16((const char*)(eattr + (size_t)e0*16) + tid*16,
                           (char*)&ea_lds[0][0] + tid*16);
  if (tid < 16) { s_lds[tid] = esrc[e0 + tid]; d_lds[tid] = edst[e0 + tid]; }

  radial_wmma<128>(embh, e0, W1f, W2f, h_lds, w_lds);
  async_wait_all();
  __syncthreads();

  #pragma unroll
  for (int it = 0; it < 4; ++it) {
    int idx = it*128 + tid;
    int el = idx >> 5, cch = idx & 31;     // one edge per wave per iteration
    int s = s_lds[el], d = d_lds[el];
    const float4* xs4 = (const float4*)(feat + (size_t)s*FDIM + cch*16);
    float4 X0 = xs4[0], X1 = xs4[1], X2 = xs4[2], X3 = xs4[3];
    float xs[16] = {X0.x,X0.y,X0.z,X0.w, X1.x,X1.y,X1.z,X1.w,
                    X2.x,X2.y,X2.z,X2.w, X3.x,X3.y,X3.z,X3.w};
    const float4* ea4 = (const float4*)&ea_lds[el][0];
    float4 E0 = ea4[0], E1 = ea4[1], E2 = ea4[2], E3 = ea4[3];
    float ea[16] = {E0.x,E0.y,E0.z,E0.w, E1.x,E1.y,E1.z,E1.w,
                    E2.x,E2.y,E2.z,E2.w, E3.x,E3.y,E3.z,E3.w};
    const float* wf = &w_lds[el][0];       // [4][32] flattened
    float m = wf[cch] * xs[0] * ea[0];
    float dt = xs[1]*ea[1] + xs[2]*ea[2] + xs[3]*ea[3];
    m += wf[32 + cch] * dt;
    dt = 0.f;
    #pragma unroll
    for (int i = 4; i < 9; ++i) dt += xs[i]*ea[i];
    m += wf[64 + cch] * dt;
    dt = 0.f;
    #pragma unroll
    for (int i = 9; i < 16; ++i) dt += xs[i]*ea[i];
    m += wf[96 + cch] * dt;
    float v = m * (INV_SQRT_MUL * INV_SQRT_NAVG);
    #pragma unroll
    for (int off = 16; off > 0; off >>= 1) v += __shfl_down(v, off, 32);
    if (lane == 0) atomicAdd(&node[d], v);
  }
}

__global__ void init_feat_kernel(const float* __restrict__ x, float* __restrict__ feat) {
  int idx = blockIdx.x * 256 + threadIdx.x;          // (n,c) pair
  if (idx >= NNODE * 32) return;
  float* p = feat + (size_t)idx * 16;
  p[0] = x[idx];
  #pragma unroll
  for (int i = 1; i < 16; ++i) p[i] = 0.f;
}

__global__ void zero_kernel(float* __restrict__ p, size_t n) {
  size_t i = (size_t)blockIdx.x * blockDim.x + threadIdx.x;
  size_t stride = (size_t)gridDim.x * blockDim.x;
  for (; i < n; i += stride) p[i] = 0.f;
}

__global__ void reduce_kernel(const float* __restrict__ node,
                              const int* __restrict__ batch,
                              float* __restrict__ out) {
  int n = blockIdx.x * 256 + threadIdx.x;
  if (n >= NNODE) return;
  atomicAdd(&out[batch[n]], node[n] * INV_SQRT_NAVG);
}

extern "C" void kernel_launch(void* const* d_in, const int* in_sizes, int n_in,
                              void* d_out, int out_size, void* d_ws, size_t ws_size,
                              hipStream_t stream) {
  (void)in_sizes; (void)n_in; (void)out_size; (void)ws_size;
  const float* pos  = (const float*)d_in[0];
  const float* x    = (const float*)d_in[1];
  const int*   bat  = (const int*)d_in[2];
  const int*   esrc = (const int*)d_in[3];
  const int*   edst = (const int*)d_in[4];
  const float* Wsc  = (const float*)d_in[5];   // [3][4][32][32]
  const float* Wr1  = (const float*)d_in[6];   // [3][10][64]
  const float* Wr2  = (const float*)d_in[7];   // [3][64][256]
  const float* Wf1  = (const float*)d_in[8];   // [10][64]
  const float* Wf2  = (const float*)d_in[9];   // [64][128]
  float* out = (float*)d_out;

  char* base = (char*)d_ws;
  size_t off = 0;
  auto wsalloc = [&](size_t bytes) -> void* {
    void* p = (void*)(base + off);
    off = (off + bytes + 255) & ~(size_t)255;
    return p;
  };
  float*    featA = (float*)wsalloc((size_t)NNODE * FDIM * 4);
  float*    featB = (float*)wsalloc((size_t)NNODE * FDIM * 4);
  float*    agg   = (float*)wsalloc((size_t)NNODE * FDIM * 4);
  float*    eattr = (float*)wsalloc((size_t)NEDGE * 16 * 4);
  _Float16* embh  = (_Float16*)wsalloc((size_t)NEDGE * 16 * 2);
  float*    node  = (float*)wsalloc((size_t)NNODE * 4);
  _Float16* W1f   = (_Float16*)wsalloc((size_t)3 * 2048 * 2);   // layer GEMM1 frags
  _Float16* W2f   = (_Float16*)wsalloc((size_t)3 * 16384 * 2);  // layer GEMM2 frags
  _Float16* Wf1f  = (_Float16*)wsalloc((size_t)2048 * 2);       // final GEMM1 frag
  _Float16* Wf2f  = (_Float16*)wsalloc((size_t)8192 * 2);       // final GEMM2 frag

  for (int L = 0; L < 3; ++L) {
    pack_w1_kernel<<<8, 256, 0, stream>>>(Wr1 + (size_t)L*640, W1f + (size_t)L*2048);
    pack_w2_kernel<<<64, 256, 0, stream>>>(Wr2 + (size_t)L*16384, W2f + (size_t)L*16384, 256);
  }
  pack_w1_kernel<<<8, 256, 0, stream>>>(Wf1, Wf1f);
  pack_w2_kernel<<<32, 256, 0, stream>>>(Wf2, Wf2f, 128);

  init_feat_kernel<<<(NNODE*32 + 255)/256, 256, 0, stream>>>(x, featA);
  geom_kernel<<<(NEDGE + 255)/256, 256, 0, stream>>>(pos, esrc, edst, eattr, embh);

  float* cur = featA; float* nxt = featB;
  for (int L = 0; L < 3; ++L) {
    zero_kernel<<<2048, 256, 0, stream>>>(agg, (size_t)NNODE * FDIM);
    layer_msg_kernel<<<NEDGE/16, 128, 0, stream>>>(cur, esrc, edst, eattr, embh,
        W1f + (size_t)L*2048, W2f + (size_t)L*16384, agg);
    sc_gate_kernel<<<NNODE/8, 256, 0, stream>>>(cur, agg, Wsc + (size_t)L*4096, nxt);
    float* t = cur; cur = nxt; nxt = t;
  }

  zero_kernel<<<64, 256, 0, stream>>>(node, (size_t)NNODE);
  final_msg_kernel<<<NEDGE/16, 128, 0, stream>>>(cur, esrc, edst, eattr, embh, Wf1f, Wf2f, node);
  zero_kernel<<<1, 32, 0, stream>>>(out, 16);
  reduce_kernel<<<(NNODE + 255)/256, 256, 0, stream>>>(node, bat, out);
}
